// HierarchicalDiffAttentionEncoderLayer_28501402976988
// MI455X (gfx1250) — matrix operand order repaired and checked by reading, not compiled, not run on hardware
//
#include <hip/hip_runtime.h>
#include <hip/hip_bf16.h>

#define B_  4
#define S_  1024
#define D_  1024
#define H_  16
#define DH_ 64

typedef __attribute__((ext_vector_type(16))) __bf16 v16bf;
typedef __attribute__((ext_vector_type(8)))  __bf16 v8bf;
typedef __attribute__((ext_vector_type(8)))  float  v8f;
typedef __attribute__((ext_vector_type(4)))  unsigned su32x4;
typedef __attribute__((ext_vector_type(8)))  unsigned su32x8;

__device__ inline __bf16 f2bf(float f) {
  unsigned u = __builtin_bit_cast(unsigned, f);
  u += 0x7FFFu + ((u >> 16) & 1u);           // round-to-nearest-even
  unsigned short h = (unsigned short)(u >> 16);
  return __builtin_bit_cast(__bf16, h);
}

// ---------------------------------------------------------------- TDM 2D tile load
// DMA a tile0 x tile1 tile of bf16 (2-byte) elements from a row-major tensor
// (row length dim0, dim1 rows, row stride dim0) into LDS at byte offset
// lds_off, adding 16B of LDS padding per 128B row chunk (=> 72-element rows
// for tile0==64). Issued wave-uniform; tracked by TENSORcnt.
__device__ inline void tdm_load_2d(unsigned lds_off, const void* gbase,
                                   long long elem_off, unsigned dim0,
                                   unsigned dim1, unsigned tile0,
                                   unsigned tile1) {
  unsigned long long ga =
      (unsigned long long)(size_t)gbase + (unsigned long long)elem_off * 2ull;
  su32x4 g0;
  g0[0] = 1u;                                    // count=1 (valid descriptor)
  g0[1] = lds_off;                               // lds_addr (bytes)
  g0[2] = (unsigned)(ga & 0xFFFFFFFFu);          // global_addr[31:0]
  g0[3] = (unsigned)((ga >> 32) & 0x1FFFFFFu)    // global_addr[56:32]
          | (2u << 30);                          // type=2 ("image")
  su32x8 g1;
  g1[0] = (1u << 16)                             // data_size: 2 bytes
          | (1u << 20)                           // pad_enable
          | (4u << 22)                           // pad_interval: 2^(4+1)=32 DW (128B)
          | (3u << 25);                          // pad_amount: 4 DWORDs (16B)
  g1[1] = (dim0 & 0xFFFFu) << 16;                // tensor_dim0[15:0]
  g1[2] = ((dim0 >> 16) & 0xFFFFu) | ((dim1 & 0xFFFFu) << 16);
  g1[3] = ((dim1 >> 16) & 0xFFFFu) | (tile0 << 16);
  g1[4] = tile1 & 0xFFFFu;                       // tile_dim1 (tile_dim2=0)
  g1[5] = dim0;                                  // tensor_dim0_stride[31:0]
  g1[6] = 0u;                                    // stride hi, dim1_stride lo
  g1[7] = 0u;
  asm volatile("tensor_load_to_lds %0, %1" :: "s"(g0), "s"(g1) : "memory");
}

// ---------------------------------------------------------------- cvt f32->bf16
__global__ void cvt_bf16_kernel(const float* __restrict__ src,
                                __bf16* __restrict__ dst, int n) {
  int i = blockIdx.x * blockDim.x + threadIdx.x;
  int stride = gridDim.x * blockDim.x;
  for (; i < n; i += stride) dst[i] = f2bf(src[i]);
}

// ---------------------------------------------------------------- bf16 WMMA GEMM
// C[M,N] = A[M,K] * W[N,K]^T + bias[N], K-contiguous row-major operands.
// TDM double-buffered LDS staging; 8 waves, 128x256 block tile, 64x64 wave tile.
__global__ __launch_bounds__(256) void gemm_bf16_kernel(
    const __bf16* __restrict__ A, const __bf16* __restrict__ W,
    const float* __restrict__ bias, float* __restrict__ C,
    int M, int N, int K) {
  constexpr int BM = 128, BN = 256, BK = 64, LSTR = 72;  // 128B row + 16B TDM pad
  __shared__ __bf16 As[2][BM * LSTR];
  __shared__ __bf16 Bs[2][BN * LSTR];

  const int t    = threadIdx.x;
  const int lane = t & 31;
  const int w    = t >> 5;
  const int half = lane >> 4;
  const int r    = lane & 15;
  const int tileN0 = blockIdx.x * BN;
  const int tileM0 = blockIdx.y * BM;
  const int mrow0  = (w >> 2) * 64;   // 2 wave-rows of 64
  const int ncol0  = (w & 3) * 64;    // 4 wave-cols of 64

  const unsigned ldsA[2] = {(unsigned)(size_t)&As[0][0], (unsigned)(size_t)&As[1][0]};
  const unsigned ldsB[2] = {(unsigned)(size_t)&Bs[0][0], (unsigned)(size_t)&Bs[1][0]};

  v8f acc[4][4];
#pragma unroll
  for (int fm = 0; fm < 4; ++fm)
#pragma unroll
    for (int fn = 0; fn < 4; ++fn)
#pragma unroll
      for (int e = 0; e < 8; ++e) acc[fm][fn][e] = 0.f;

  const int nT = K / BK;
  if (w == 0) {
    tdm_load_2d(ldsA[0], A, (long long)tileM0 * K, K, M, BK, BM);
    tdm_load_2d(ldsB[0], W, (long long)tileN0 * K, K, N, BK, BN);
  }
  for (int it = 0; it < nT; ++it) {
    const int cur = it & 1;
    if (w == 0) {
      if (it + 1 < nT) {
        long long k0n = (long long)(it + 1) * BK;
        tdm_load_2d(ldsA[cur ^ 1], A, (long long)tileM0 * K + k0n, K, M, BK, BM);
        tdm_load_2d(ldsB[cur ^ 1], W, (long long)tileN0 * K + k0n, K, N, BK, BN);
        __builtin_amdgcn_s_wait_tensorcnt(2);   // previous stage complete
      } else {
        __builtin_amdgcn_s_wait_tensorcnt(0);
      }
    }
    __syncthreads();
    const __bf16* Asc = As[cur];
    const __bf16* Bsc = Bs[cur];
#pragma unroll
    for (int kk = 0; kk < 2; ++kk) {
      v16bf af[4], bfr[4];
#pragma unroll
      for (int fm = 0; fm < 4; ++fm) {
        const __bf16* p = &Asc[(mrow0 + fm * 16 + r) * LSTR + kk * 32 + half * 8];
        v8bf lo = *(const v8bf*)p;
        v8bf hi = *(const v8bf*)(p + 16);
#pragma unroll
        for (int e = 0; e < 8; ++e) { af[fm][e] = lo[e]; af[fm][e + 8] = hi[e]; }
      }
#pragma unroll
      for (int fn = 0; fn < 4; ++fn) {
        const __bf16* p = &Bsc[(ncol0 + fn * 16 + r) * LSTR + kk * 32 + half * 8];
        v8bf lo = *(const v8bf*)p;
        v8bf hi = *(const v8bf*)(p + 16);
#pragma unroll
        for (int e = 0; e < 8; ++e) { bfr[fn][e] = lo[e]; bfr[fn][e + 8] = hi[e]; }
      }
#pragma unroll
      for (int fm = 0; fm < 4; ++fm)
#pragma unroll
        for (int fn = 0; fn < 4; ++fn)
          acc[fm][fn] = __builtin_amdgcn_wmma_f32_16x16x32_bf16(
              false, af[fm], false, bfr[fn], (short)0, acc[fm][fn], false, false);
    }
    __syncthreads();
  }
#pragma unroll
  for (int fm = 0; fm < 4; ++fm)
#pragma unroll
    for (int fn = 0; fn < 4; ++fn) {
      int col  = tileN0 + ncol0 + fn * 16 + r;
      int row0 = tileM0 + mrow0 + fm * 16 + half * 8;
      float bv = bias[col];
#pragma unroll
      for (int e = 0; e < 8; ++e)
        C[(size_t)(row0 + e) * N + col] = acc[fm][fn][e] + bv;
    }
}

// ---------------------------------------------------------------- logL cumsum
__global__ void logl_kernel(const float* __restrict__ a, float* __restrict__ logL) {
  int b = threadIdx.x;
  if (b >= B_) return;
  float acc = 0.f;
  logL[(size_t)b * S_] = 0.f;
  for (int s = 1; s < S_; ++s) {
    acc += __logf(a[(size_t)b * (S_ - 1) + s - 1]);
    logL[(size_t)b * S_ + s] = acc;
  }
}

// ---------------------------------------------------------------- RoPE + repack
__global__ void rope_pack_kernel(const float* __restrict__ qp, const float* __restrict__ kp,
                                 const float* __restrict__ vp,
                                 __bf16* __restrict__ q1r, __bf16* __restrict__ q2r,
                                 __bf16* __restrict__ k1r, __bf16* __restrict__ k2r,
                                 __bf16* __restrict__ vb) {
  int idx = blockIdx.x * blockDim.x + threadIdx.x;   // B*H*S*(DH/2)
  if (idx >= B_ * H_ * S_ * (DH_ / 2)) return;
  int j = idx & 31;
  int s = (idx >> 5) & (S_ - 1);
  int h = (idx >> 15) & (H_ - 1);
  int b = idx >> 19;

  float fr = __powf(10000.0f, -((float)(2 * j) / (float)DH_));
  float ang = (float)s * fr;
  float sn, cs;
  __sincosf(ang, &sn, &cs);

  size_t rowq = ((size_t)b * S_ + s) * (2 * D_);
  size_t rowv = ((size_t)b * S_ + s) * D_;
  size_t o1   = (size_t)h * DH_;
  size_t orow = (((size_t)b * H_ + h) * S_ + s) * DH_;

  float x1 = qp[rowq + o1 + j], x2 = qp[rowq + o1 + j + 32];
  q1r[orow + j]      = f2bf(x1 * cs - x2 * sn);
  q1r[orow + j + 32] = f2bf(x1 * sn + x2 * cs);
  x1 = qp[rowq + D_ + o1 + j]; x2 = qp[rowq + D_ + o1 + j + 32];
  q2r[orow + j]      = f2bf(x1 * cs - x2 * sn);
  q2r[orow + j + 32] = f2bf(x1 * sn + x2 * cs);
  x1 = kp[rowq + o1 + j]; x2 = kp[rowq + o1 + j + 32];
  k1r[orow + j]      = f2bf(x1 * cs - x2 * sn);
  k1r[orow + j + 32] = f2bf(x1 * sn + x2 * cs);
  x1 = kp[rowq + D_ + o1 + j]; x2 = kp[rowq + D_ + o1 + j + 32];
  k2r[orow + j]      = f2bf(x1 * cs - x2 * sn);
  k2r[orow + j + 32] = f2bf(x1 * sn + x2 * cs);
  vb[orow + j]       = f2bf(vp[rowv + o1 + j]);
  vb[orow + j + 32]  = f2bf(vp[rowv + o1 + j + 32]);
}

// ---------------------------------------------------------------- fused diff-attention
__global__ __launch_bounds__(128) void attn_kernel(
    const __bf16* __restrict__ q1r, const __bf16* __restrict__ q2r,
    const __bf16* __restrict__ k1r, const __bf16* __restrict__ k2r,
    const __bf16* __restrict__ vb, const float* __restrict__ logL,
    const float* __restrict__ lam_p, float* __restrict__ ctxb) {
  constexpr int STR = 72;
  __shared__ __bf16 K1s[2][64 * STR];   // TDM double-buffered
  __shared__ __bf16 K2s[2][64 * STR];
  __shared__ __bf16 Vst[64 * STR];      // transposed: [d][j]
  __shared__ __bf16 Ps1[64 * STR];      // wave-private 16-row slices
  __shared__ __bf16 Ps2[64 * STR];
  __shared__ float  logLjs[64];

  const int t    = threadIdx.x;
  const int lane = t & 31;
  const int w    = t >> 5;
  const int half = lane >> 4;
  const int r    = lane & 15;

  int gid = blockIdx.x;
  int ib  = gid & 15;
  int h   = (gid >> 4) & (H_ - 1);
  int b   = gid >> 8;
  int i0  = ib * 64;
  size_t bh = (size_t)b * H_ + h;

  float lamv = lam_p[0];
  const float sc = 0.125f;  // 1/sqrt(DH)
  const unsigned ldsK1[2] = {(unsigned)(size_t)&K1s[0][0], (unsigned)(size_t)&K1s[1][0]};
  const unsigned ldsK2[2] = {(unsigned)(size_t)&K2s[0][0], (unsigned)(size_t)&K2s[1][0]};

  // preload Q fragments (A-matrix layout, two K-steps of 32)
  v16bf q1f[2], q2f[2];
  {
    const __bf16* qrow1 = q1r + (bh * S_ + i0 + w * 16 + r) * DH_;
    const __bf16* qrow2 = q2r + (bh * S_ + i0 + w * 16 + r) * DH_;
#pragma unroll
    for (int kk = 0; kk < 2; ++kk) {
      int kb = kk * 32 + half * 8;
      v8bf lo = *(const v8bf*)(qrow1 + kb);
      v8bf hi = *(const v8bf*)(qrow1 + kb + 16);
#pragma unroll
      for (int e = 0; e < 8; ++e) { q1f[kk][e] = lo[e]; q1f[kk][e + 8] = hi[e]; }
      lo = *(const v8bf*)(qrow2 + kb);
      hi = *(const v8bf*)(qrow2 + kb + 16);
#pragma unroll
      for (int e = 0; e < 8; ++e) { q2f[kk][e] = lo[e]; q2f[kk][e + 8] = hi[e]; }
    }
  }
  float logLi[8];
#pragma unroll
  for (int e = 0; e < 8; ++e)
    logLi[e] = logL[(size_t)b * S_ + i0 + w * 16 + e + half * 8];

  float m1[8], l1[8], m2[8], l2[8];
  v8f ctx1[4], ctx2[4];
#pragma unroll
  for (int e = 0; e < 8; ++e) { m1[e] = -1e30f; l1[e] = 0.f; m2[e] = -1e30f; l2[e] = 0.f; }
#pragma unroll
  for (int fn = 0; fn < 4; ++fn)
#pragma unroll
    for (int e = 0; e < 8; ++e) { ctx1[fn][e] = 0.f; ctx2[fn][e] = 0.f; }

  __bf16* Pw1 = &Ps1[w * 16 * STR];
  __bf16* Pw2 = &Ps2[w * 16 * STR];

  const int nJ = S_ / 64;
  if (w == 0) {
    long long ro = (long long)(bh * S_) * DH_;
    tdm_load_2d(ldsK1[0], k1r, ro, DH_, B_ * H_ * S_, 64, 64);
    tdm_load_2d(ldsK2[0], k2r, ro, DH_, B_ * H_ * S_, 64, 64);
  }
  for (int jb = 0; jb < nJ; ++jb) {
    const int cur = jb & 1;
    // TDM double-buffer: stage next K1/K2 tiles, wait for current
    if (w == 0) {
      if (jb + 1 < nJ) {
        long long ro = (long long)(bh * S_ + (size_t)(jb + 1) * 64) * DH_;
        tdm_load_2d(ldsK1[cur ^ 1], k1r, ro, DH_, B_ * H_ * S_, 64, 64);
        tdm_load_2d(ldsK2[cur ^ 1], k2r, ro, DH_, B_ * H_ * S_, 64, 64);
        __builtin_amdgcn_s_wait_tensorcnt(2);
      } else {
        __builtin_amdgcn_s_wait_tensorcnt(0);
      }
    }
    // cooperative transposed stage of V tile
#pragma unroll
    for (int q = 0; q < 4; ++q) {
      int cc  = t + 128 * q;
      int row = cc >> 3;
      int off = (cc & 7) * 8;
      size_t g = (bh * S_ + (size_t)jb * 64 + row) * DH_ + off;
      v8bf vv = *(const v8bf*)&vb[g];
#pragma unroll
      for (int e = 0; e < 8; ++e) Vst[(off + e) * STR + row] = vv[e];
    }
    if (t < 64) logLjs[t] = logL[(size_t)b * S_ + jb * 64 + t];
    if (jb + 1 < nJ) {  // prefetch next V tile (global_prefetch_b8)
      const __bf16* nv = &vb[(bh * S_ + (size_t)(jb + 1) * 64) * DH_ + (size_t)t * 32];
      __builtin_prefetch((const void*)nv, 0, 0);
    }
    __syncthreads();
    const __bf16* K1c = K1s[cur];
    const __bf16* K2c = K2s[cur];

    // scores S1 = Q1*K1^T, S2 = Q2*K2^T  (16 i-rows x 64 j)
    v8f s1[4], s2[4];
#pragma unroll
    for (int fn = 0; fn < 4; ++fn)
#pragma unroll
      for (int e = 0; e < 8; ++e) { s1[fn][e] = 0.f; s2[fn][e] = 0.f; }
#pragma unroll
    for (int kk = 0; kk < 2; ++kk) {
#pragma unroll
      for (int fn = 0; fn < 4; ++fn) {
        const __bf16* p1 = &K1c[(fn * 16 + r) * STR + kk * 32 + half * 8];
        v8bf lo = *(const v8bf*)p1; v8bf hi = *(const v8bf*)(p1 + 16);
        v16bf bfr;
#pragma unroll
        for (int e = 0; e < 8; ++e) { bfr[e] = lo[e]; bfr[e + 8] = hi[e]; }
        s1[fn] = __builtin_amdgcn_wmma_f32_16x16x32_bf16(
            false, q1f[kk], false, bfr, (short)0, s1[fn], false, false);
        const __bf16* p2 = &K2c[(fn * 16 + r) * STR + kk * 32 + half * 8];
        lo = *(const v8bf*)p2; hi = *(const v8bf*)(p2 + 16);
#pragma unroll
        for (int e = 0; e < 8; ++e) { bfr[e] = lo[e]; bfr[e + 8] = hi[e]; }
        s2[fn] = __builtin_amdgcn_wmma_f32_16x16x32_bf16(
            false, q2f[kk], false, bfr, (short)0, s2[fn], false, false);
      }
    }
#pragma unroll
    for (int fn = 0; fn < 4; ++fn)
#pragma unroll
      for (int e = 0; e < 8; ++e) { s1[fn][e] *= sc; s2[fn][e] *= sc; }

    // row max over 64 j (4 frags elementwise, then xor-shuffle within lane-half)
    float rm1[8], rm2[8];
#pragma unroll
    for (int e = 0; e < 8; ++e) {
      float a1v = fmaxf(fmaxf(s1[0][e], s1[1][e]), fmaxf(s1[2][e], s1[3][e]));
      float a2v = fmaxf(fmaxf(s2[0][e], s2[1][e]), fmaxf(s2[2][e], s2[3][e]));
#pragma unroll
      for (int md = 1; md < 16; md <<= 1) {
        a1v = fmaxf(a1v, __shfl_xor(a1v, md, 32));
        a2v = fmaxf(a2v, __shfl_xor(a2v, md, 32));
      }
      rm1[e] = a1v; rm2[e] = a2v;
    }
    // online softmax rescale
#pragma unroll
    for (int e = 0; e < 8; ++e) {
      float mn = fmaxf(m1[e], rm1[e]);
      float c  = __expf(m1[e] - mn);
      m1[e] = mn; l1[e] *= c;
#pragma unroll
      for (int fn = 0; fn < 4; ++fn) ctx1[fn][e] *= c;
      mn = fmaxf(m2[e], rm2[e]);
      c  = __expf(m2[e] - mn);
      m2[e] = mn; l2[e] *= c;
#pragma unroll
      for (int fn = 0; fn < 4; ++fn) ctx2[fn][e] *= c;
    }
    // P = exp(s - m), C-mask folded in; write bf16 P tiles (wave-private LDS)
    float ps1[8], ps2[8];
#pragma unroll
    for (int e = 0; e < 8; ++e) { ps1[e] = 0.f; ps2[e] = 0.f; }
#pragma unroll
    for (int fn = 0; fn < 4; ++fn) {
      int jl = fn * 16 + r;
      float clj = logLjs[jl];
#pragma unroll
      for (int e = 0; e < 8; ++e) {
        float ci = __expf(-fabsf(logLi[e] - clj));
        float p1 = __expf(s1[fn][e] - m1[e]); ps1[e] += p1;
        float p2 = __expf(s2[fn][e] - m2[e]); ps2[e] += p2;
        Pw1[(e + half * 8) * STR + jl] = f2bf(p1 * ci);
        Pw2[(e + half * 8) * STR + jl] = f2bf(p2 * ci);
      }
    }
#pragma unroll
    for (int e = 0; e < 8; ++e) {
      float a1v = ps1[e], a2v = ps2[e];
#pragma unroll
      for (int md = 1; md < 16; md <<= 1) {
        a1v += __shfl_xor(a1v, md, 32);
        a2v += __shfl_xor(a2v, md, 32);
      }
      l1[e] += a1v; l2[e] += a2v;
    }
    // ctx += (P .* C) @ V via WMMA (two K-steps over j)
#pragma unroll
    for (int kk = 0; kk < 2; ++kk) {
      int kb = kk * 32 + half * 8;
      const __bf16* pa1 = &Pw1[r * STR + kb];
      v8bf lo = *(const v8bf*)pa1; v8bf hi = *(const v8bf*)(pa1 + 16);
      v16bf a1f;
#pragma unroll
      for (int e = 0; e < 8; ++e) { a1f[e] = lo[e]; a1f[e + 8] = hi[e]; }
      const __bf16* pa2 = &Pw2[r * STR + kb];
      lo = *(const v8bf*)pa2; hi = *(const v8bf*)(pa2 + 16);
      v16bf a2f;
#pragma unroll
      for (int e = 0; e < 8; ++e) { a2f[e] = lo[e]; a2f[e + 8] = hi[e]; }
#pragma unroll
      for (int fn = 0; fn < 4; ++fn) {
        const __bf16* pv = &Vst[(fn * 16 + r) * STR + kk * 32 + half * 8];
        v8bf lv = *(const v8bf*)pv; v8bf hv = *(const v8bf*)(pv + 16);
        v16bf bvf;
#pragma unroll
        for (int e = 0; e < 8; ++e) { bvf[e] = lv[e]; bvf[e + 8] = hv[e]; }
        ctx1[fn] = __builtin_amdgcn_wmma_f32_16x16x32_bf16(
            false, a1f, false, bvf, (short)0, ctx1[fn], false, false);
        ctx2[fn] = __builtin_amdgcn_wmma_f32_16x16x32_bf16(
            false, a2f, false, bvf, (short)0, ctx2[fn], false, false);
      }
    }
    __syncthreads();
  }
  // ctx = acc1/l1 - lam * acc2/l2  -> (B,H,S,DH) f32
#pragma unroll
  for (int fn = 0; fn < 4; ++fn)
#pragma unroll
    for (int e = 0; e < 8; ++e) {
      float o = ctx1[fn][e] / l1[e] - lamv * (ctx2[fn][e] / l2[e]);
      ctxb[(bh * S_ + i0 + w * 16 + e + half * 8) * DH_ + fn * 16 + r] = o;
    }
}

// ---------------------------------------------------------------- GroupNorm stats
__global__ __launch_bounds__(256) void gn_stats_kernel(const float* __restrict__ ctxb,
                                                       float* __restrict__ stats) {
  int bh = blockIdx.x;
  const float* p = ctxb + (size_t)bh * S_ * DH_;
  float s = 0.f, s2 = 0.f;
  for (int i = threadIdx.x; i < S_ * DH_; i += 256) {
    float v = p[i]; s += v; s2 += v * v;
  }
  __shared__ float sh[256], sh2[256];
  sh[threadIdx.x] = s; sh2[threadIdx.x] = s2;
  __syncthreads();
  for (int st = 128; st > 0; st >>= 1) {
    if (threadIdx.x < st) {
      sh[threadIdx.x]  += sh[threadIdx.x + st];
      sh2[threadIdx.x] += sh2[threadIdx.x + st];
    }
    __syncthreads();
  }
  if (threadIdx.x == 0) {
    float inv = 1.0f / (float)(S_ * DH_);
    float mean = sh[0] * inv;
    float var  = sh2[0] * inv - mean * mean;
    stats[bh * 2] = mean; stats[bh * 2 + 1] = var;
  }
}

// ---------------------------------------------------------------- GN apply + pack bf16
__global__ void gn_apply_kernel(const float* __restrict__ ctxb, const float* __restrict__ stats,
                                const float* __restrict__ gnw, const float* __restrict__ gnb,
                                __bf16* __restrict__ xn2) {
  int idx = blockIdx.x * blockDim.x + threadIdx.x;
  if (idx >= B_ * H_ * S_ * DH_) return;
  int d  = idx & (DH_ - 1);
  int s  = (idx >> 6) & (S_ - 1);
  int bh = idx >> 16;
  int h  = bh & (H_ - 1);
  int b  = bh >> 4;
  float mean = stats[bh * 2], var = stats[bh * 2 + 1];
  float v = (ctxb[idx] - mean) * rsqrtf(var + 1e-5f);
  v = v * gnw[h * DH_ + d] + gnb[h * DH_ + d];
  v *= 0.8f;  // 1 - LAMBDA_INIT
  xn2[((size_t)b * S_ + s) * D_ + h * DH_ + d] = f2bf(v);
}

// ---------------------------------------------------------------- launcher
extern "C" void kernel_launch(void* const* d_in, const int* in_sizes, int n_in,
                              void* d_out, int out_size, void* d_ws, size_t ws_size,
                              hipStream_t stream) {
  const float* x_norm = (const float*)d_in[0];
  const float* aff    = (const float*)d_in[1];
  const float* Wq = (const float*)d_in[2];
  const float* bq = (const float*)d_in[3];
  const float* Wk = (const float*)d_in[4];
  const float* bk = (const float*)d_in[5];
  const float* Wv = (const float*)d_in[6];
  const float* bv = (const float*)d_in[7];
  const float* Wo = (const float*)d_in[8];
  const float* bo = (const float*)d_in[9];
  const float* gnw = (const float*)d_in[10];
  const float* gnb = (const float*)d_in[11];
  const float* lam = (const float*)d_in[12];
  float* out = (float*)d_out;

  char* ws = (char*)d_ws;
  size_t off = 0;
  auto alloc = [&](size_t bytes) -> void* {
    void* p = ws + off;
    off = (off + bytes + 255) & ~(size_t)255;
    return p;
  };

  const size_t M = (size_t)B_ * S_;  // 4096
  __bf16* xb   = (__bf16*)alloc(M * D_ * 2);
  __bf16* wqb  = (__bf16*)alloc((size_t)2 * D_ * D_ * 2);
  __bf16* wkb  = (__bf16*)alloc((size_t)2 * D_ * D_ * 2);
  __bf16* wvb  = (__bf16*)alloc((size_t)D_ * D_ * 2);
  __bf16* wob  = (__bf16*)alloc((size_t)D_ * D_ * 2);
  float*  qp   = (float*)alloc(M * 2 * D_ * 4);
  float*  kp   = (float*)alloc(M * 2 * D_ * 4);
  float*  vp   = (float*)alloc(M * D_ * 4);
  __bf16* q1r  = (__bf16*)alloc((size_t)B_ * H_ * S_ * DH_ * 2);
  __bf16* q2r  = (__bf16*)alloc((size_t)B_ * H_ * S_ * DH_ * 2);
  __bf16* k1r  = (__bf16*)alloc((size_t)B_ * H_ * S_ * DH_ * 2);
  __bf16* k2r  = (__bf16*)alloc((size_t)B_ * H_ * S_ * DH_ * 2);
  __bf16* vbp  = (__bf16*)alloc((size_t)B_ * H_ * S_ * DH_ * 2);
  float*  logL = (float*)alloc((size_t)B_ * S_ * 4);
  float*  ctxb = (float*)alloc((size_t)B_ * H_ * S_ * DH_ * 4);
  __bf16* xn2  = (__bf16*)alloc(M * D_ * 2);
  float*  stats = (float*)alloc((size_t)B_ * H_ * 2 * 4);

  // 1. bf16 conversions
  cvt_bf16_kernel<<<1024, 256, 0, stream>>>(x_norm, xb, B_ * S_ * D_);
  cvt_bf16_kernel<<<1024, 256, 0, stream>>>(Wq, wqb, 2 * D_ * D_);
  cvt_bf16_kernel<<<1024, 256, 0, stream>>>(Wk, wkb, 2 * D_ * D_);
  cvt_bf16_kernel<<<1024, 256, 0, stream>>>(Wv, wvb, D_ * D_);
  cvt_bf16_kernel<<<1024, 256, 0, stream>>>(Wo, wob, D_ * D_);

  // 2. projections
  gemm_bf16_kernel<<<dim3((2 * D_) / 256, M / 128), 256, 0, stream>>>(
      xb, wqb, bq, qp, (int)M, 2 * D_, D_);
  gemm_bf16_kernel<<<dim3((2 * D_) / 256, M / 128), 256, 0, stream>>>(
      xb, wkb, bk, kp, (int)M, 2 * D_, D_);
  gemm_bf16_kernel<<<dim3(D_ / 256, M / 128), 256, 0, stream>>>(
      xb, wvb, bv, vp, (int)M, D_, D_);

  // 3. hierarchical-mask log-prefix + RoPE/head-major bf16 repack
  logl_kernel<<<1, 32, 0, stream>>>(aff, logL);
  {
    int n = B_ * H_ * S_ * (DH_ / 2);
    rope_pack_kernel<<<(n + 255) / 256, 256, 0, stream>>>(qp, kp, vp, q1r, q2r, k1r, k2r, vbp);
  }

  // 4. fused differential attention
  attn_kernel<<<B_ * H_ * (S_ / 64), 128, 0, stream>>>(q1r, q2r, k1r, k2r, vbp,
                                                       logL, lam, ctxb);

  // 5. group norm + scale + bf16 pack
  gn_stats_kernel<<<B_ * H_, 256, 0, stream>>>(ctxb, stats);
  {
    int n = B_ * H_ * S_ * DH_;
    gn_apply_kernel<<<(n + 255) / 256, 256, 0, stream>>>(ctxb, stats, gnw, gnb, xn2);
  }

  // 6. output projection -> d_out (f32)
  gemm_bf16_kernel<<<dim3(D_ / 256, M / 128), 256, 0, stream>>>(
      xn2, wob, bo, out, (int)M, D_, D_);

  (void)in_sizes; (void)n_in; (void)out_size; (void)ws_size;
}
